// GNN_Layer_2422361555230
// MI455X (gfx1250) — compile-verified, matrix operand order checked by reading, not run on hardware
//
#include <hip/hip_runtime.h>
#include <hip/hip_bf16.h>

// ---------------------------------------------------------------------------
// GNN layer:  y_e = relu([H[src] | Xe] @ W1)          (edge MLP, K=192)
//             S   = segment_max(y_e, dst)  (>=0, zero-fill)
//             out = relu([H | S] @ W2) + H            (node MLP, K=256)
// bf16 WMMA (v_wmma_f32_16x16x32_bf16), fp32 accumulate.
// Fragment traffic as B128; kt-outer loop: 16-load clause + 8 WMMAs per step.
// ---------------------------------------------------------------------------

typedef __attribute__((ext_vector_type(16))) __bf16    v16bf;
typedef __attribute__((ext_vector_type(8)))  float     v8f;
typedef __attribute__((ext_vector_type(4)))  unsigned  u32x4;
typedef __attribute__((ext_vector_type(4)))  float     f32x4;

#define D_H   128
#define D_E   64
#define HID   128
#define K1    (D_H + D_E)    // 192
#define K2    (D_H + HID)    // 256

// round-to-nearest-even fp32 -> bf16 pair packed into one dword
__device__ __forceinline__ unsigned pack2_bf16(float a, float b) {
    unsigned ua = __float_as_uint(a), ub = __float_as_uint(b);
    ua = (ua + 0x7FFFu + ((ua >> 16) & 1u)) >> 16;
    ub = (ub + 0x7FFFu + ((ub >> 16) & 1u)) >> 16;
    return ua | (ub << 16);
}

__device__ __forceinline__ u32x4 pack8_bf16(f32x4 a, f32x4 b) {
    u32x4 q;
    q.x = pack2_bf16(a.x, a.y);
    q.y = pack2_bf16(a.z, a.w);
    q.z = pack2_bf16(b.x, b.y);
    q.w = pack2_bf16(b.z, b.w);
    return q;
}

// 16-bit WMMA A/B fragment: dword v holds K = kb + (v&4)*4 + hi*8 + (v&3)*2 {+0,+1}
// => q[0] = 8 consecutive bf16 at K = kb + hi*8, q[1] = same at +16.  (§7.12.2)
union AB16 { u32x4 q[2]; v16bf v; };

// ---------------------------------------------------------------------------
// Weight conversion: W[k][n] fp32 row-major  ->  Wt[n][k] bf16 (K contiguous)
// ---------------------------------------------------------------------------
__global__ void cvt_w_kernel(const float* __restrict__ W, unsigned* __restrict__ Wt,
                             int K, int N) {
    int i = blockIdx.x * blockDim.x + threadIdx.x;      // one dword = 2 k's
    int total = N * (K >> 1);
    if (i >= total) return;
    int n  = i / (K >> 1);
    int k2 = i - n * (K >> 1);
    Wt[i] = pack2_bf16(W[(2 * k2 + 0) * N + n], W[(2 * k2 + 1) * N + n]);
}

// ---------------------------------------------------------------------------
// Edge kernel: 128-edge tile, 8 waves, each wave = 16-edge M-strip x full N=128
// ---------------------------------------------------------------------------
__global__ __launch_bounds__(256)
void edge_mlp_scatter_kernel(const float* __restrict__ H,
                             const int*   __restrict__ src,
                             const int*   __restrict__ dst,
                             const float* __restrict__ Xe,
                             const u32x4* __restrict__ W1q,   // bf16 [128][192]
                             int* __restrict__ S,             // fp32 bits, >=0
                             int nEdges) {
    __shared__ u32x4 ldsA[128 * (K1 / 8)];             // 128 x 192 bf16 = 48KB
    const int RQ = K1 / 8;                             // 24 u32x4 per row

    const int tid  = threadIdx.x;
    const int base = blockIdx.x * 128;

    // ---- cooperative gather + bf16 pack into LDS (B128 in, B128 out) ----
    const f32x4* H4  = (const f32x4*)H;
    const f32x4* Xe4 = (const f32x4*)Xe;
    for (int i = tid; i < 128 * RQ; i += 256) {
        int r = i / RQ;
        int g = i - r * RQ;                            // 8-float group in row
        int e  = base + r;
        int ec = e < nEdges ? e : nEdges - 1;
        f32x4 fa, fb;
        if (g < D_H / 8) {
            int s = src[ec];
            fa = H4[s * (D_H / 4) + 2 * g];
            fb = H4[s * (D_H / 4) + 2 * g + 1];
        } else {
            int k = g - D_H / 8;
            fa = Xe4[ec * (D_E / 4) + 2 * k];
            fb = Xe4[ec * (D_E / 4) + 2 * k + 1];
        }
        ldsA[i] = pack8_bf16(fa, fb);
    }
    __syncthreads();

    const int w    = tid >> 5;
    const int lane = tid & 31;
    const int lo   = lane & 15;
    const int hi   = lane >> 4;

    // ---- hoist all A fragments for this wave's 16-edge strip ----
    const int rowQ = (w * 16 + lo) * RQ;
    AB16 a[K1 / 32];
    #pragma unroll
    for (int kt = 0; kt < K1 / 32; ++kt) {
        a[kt].q[0] = ldsA[rowQ + 4 * kt + hi];
        a[kt].q[1] = ldsA[rowQ + 4 * kt + hi + 2];
    }

    v8f acc[8] = {};                                   // 8 N-tiles of 16x16 f32
    #pragma unroll
    for (int kt = 0; kt < K1 / 32; ++kt) {             // kt-outer
        AB16 b[8];
        #pragma unroll
        for (int nt = 0; nt < 8; ++nt) {               // 16 b128 loads (clause)
            const int n = nt * 16 + lo;
            b[nt].q[0] = W1q[n * RQ + 4 * kt + hi];
            b[nt].q[1] = W1q[n * RQ + 4 * kt + hi + 2];
        }
        #pragma unroll
        for (int nt = 0; nt < 8; ++nt)                 // 8 back-to-back WMMAs
            acc[nt] = __builtin_amdgcn_wmma_f32_16x16x32_bf16(
                false, a[kt].v, false, b[nt].v, (short)0, acc[nt], false, false);
    }

    // ---- relu + scatter-max (non-negative floats: int max == float max) ----
    #pragma unroll
    for (int r = 0; r < 8; ++r) {
        const int e = base + w * 16 + r + 8 * hi;
        if (e < nEdges) {
            const int d = dst[e];
            #pragma unroll
            for (int nt = 0; nt < 8; ++nt) {
                float val = acc[nt][r];
                if (val > 0.0f)
                    atomicMax(&S[d * HID + nt * 16 + lo], __float_as_int(val));
            }
        }
    }
}

// ---------------------------------------------------------------------------
// Node kernel: 64-node tile, 4 waves, K=256 = [H | S], epilogue relu + H
// ---------------------------------------------------------------------------
__global__ __launch_bounds__(128)
void node_mlp_kernel(const float* __restrict__ H,
                     const float* __restrict__ S,
                     const u32x4* __restrict__ W2q,     // bf16 [128][256]
                     float* __restrict__ out,
                     int nNodes) {
    __shared__ u32x4 ldsA[64 * (K2 / 8)];              // 64 x 256 bf16 = 32KB
    const int RQ = K2 / 8;                             // 32 u32x4 per row

    const int tid  = threadIdx.x;
    const int base = blockIdx.x * 64;

    const f32x4* H4 = (const f32x4*)H;
    const f32x4* S4 = (const f32x4*)S;
    for (int i = tid; i < 64 * RQ; i += 128) {
        int r = i / RQ;
        int g = i - r * RQ;
        int n  = base + r;
        int nc = n < nNodes ? n : nNodes - 1;
        f32x4 fa, fb;
        if (g < D_H / 8) {
            fa = H4[nc * (D_H / 4) + 2 * g];
            fb = H4[nc * (D_H / 4) + 2 * g + 1];
        } else {
            int k = g - D_H / 8;
            fa = S4[nc * (HID / 4) + 2 * k];
            fb = S4[nc * (HID / 4) + 2 * k + 1];
        }
        ldsA[i] = pack8_bf16(fa, fb);
    }
    __syncthreads();

    const int w    = tid >> 5;
    const int lane = tid & 31;
    const int lo   = lane & 15;
    const int hi   = lane >> 4;

    const int rowQ = (w * 16 + lo) * RQ;
    AB16 a[K2 / 32];
    #pragma unroll
    for (int kt = 0; kt < K2 / 32; ++kt) {
        a[kt].q[0] = ldsA[rowQ + 4 * kt + hi];
        a[kt].q[1] = ldsA[rowQ + 4 * kt + hi + 2];
    }

    v8f acc[8] = {};
    #pragma unroll
    for (int kt = 0; kt < K2 / 32; ++kt) {             // kt-outer
        AB16 b[8];
        #pragma unroll
        for (int nt = 0; nt < 8; ++nt) {               // 16 b128 loads (clause)
            const int n = nt * 16 + lo;
            b[nt].q[0] = W2q[n * RQ + 4 * kt + hi];
            b[nt].q[1] = W2q[n * RQ + 4 * kt + hi + 2];
        }
        #pragma unroll
        for (int nt = 0; nt < 8; ++nt)                 // 8 back-to-back WMMAs
            acc[nt] = __builtin_amdgcn_wmma_f32_16x16x32_bf16(
                false, a[kt].v, false, b[nt].v, (short)0, acc[nt], false, false);
    }

    // ---- relu + residual ----
    #pragma unroll
    for (int r = 0; r < 8; ++r) {
        const int node = base + w * 16 + r + 8 * hi;
        if (node < nNodes) {
            #pragma unroll
            for (int nt = 0; nt < 8; ++nt) {
                const int j = nt * 16 + lo;
                out[node * HID + j] = fmaxf(acc[nt][r], 0.0f) + H[node * D_H + j];
            }
        }
    }
}

// ---------------------------------------------------------------------------
extern "C" void kernel_launch(void* const* d_in, const int* in_sizes, int n_in,
                              void* d_out, int out_size, void* d_ws, size_t ws_size,
                              hipStream_t stream) {
    const float* H   = (const float*)d_in[0];
    const int*   idx = (const int*)  d_in[1];
    const float* Xe  = (const float*)d_in[2];
    const float* W1  = (const float*)d_in[3];
    const float* W2  = (const float*)d_in[4];
    float*       out = (float*)d_out;

    const int nNodes = in_sizes[0] / D_H;
    const int nEdges = in_sizes[2] / D_E;
    const int* src = idx;
    const int* dst = idx + nEdges;

    // workspace layout: S (fp32 N x HID) | W1t bf16 | W2t bf16
    char* ws = (char*)d_ws;
    size_t s_bytes = (size_t)nNodes * HID * sizeof(float);
    int*      Sbuf = (int*)ws;
    unsigned* W1t  = (unsigned*)(ws + s_bytes);
    unsigned* W2t  = W1t + (HID * K1) / 2;

    // 1) bf16 K-major weight copies (tiny, L2-resident afterwards)
    cvt_w_kernel<<<(HID * K1 / 2 + 255) / 256, 256, 0, stream>>>(W1, W1t, K1, HID);
    cvt_w_kernel<<<(HID * K2 / 2 + 255) / 256, 256, 0, stream>>>(W2, W2t, K2, HID);

    // 2) zero scatter-max target (zero == relu'd empty-segment fill)
    hipMemsetAsync(Sbuf, 0, s_bytes, stream);

    // 3) edge MLP + scatter-max
    edge_mlp_scatter_kernel<<<(nEdges + 127) / 128, 256, 0, stream>>>(
        H, src, dst, Xe, (const u32x4*)W1t, Sbuf, nEdges);

    // 4) node MLP + residual
    node_mlp_kernel<<<(nNodes + 63) / 64, 128, 0, stream>>>(
        H, (const float*)Sbuf, (const u32x4*)W2t, out, nNodes);
}
